// SingleHeadAttention_78254304133141
// MI455X (gfx1250) — matrix operand order repaired and checked by reading, not compiled, not run on hardware
//
#include <hip/hip_runtime.h>
#include <hip/hip_bf16.h>

// Problem constants (match reference).
#define EMB  1024
#define HEAD 128
#define TSEQ 2048
#define NB   8
#define NT   (HEAD / 16)   // 8 N-tiles over the head dim

typedef __attribute__((ext_vector_type(16))) __bf16   v16bf;
typedef __attribute__((ext_vector_type(8)))  __bf16   v8bf;
typedef __attribute__((ext_vector_type(8)))  float    v8f;
typedef __attribute__((ext_vector_type(4)))  unsigned v4u;
typedef __attribute__((ext_vector_type(4)))  int      v4i;
typedef __attribute__((ext_vector_type(8)))  int      v8i;

#if defined(__gfx1250__) && __has_builtin(__builtin_amdgcn_tensor_load_to_lds) && \
    __has_builtin(__builtin_amdgcn_s_wait_tensorcnt)
#define USE_TDM 1
#endif

__device__ __forceinline__ __bf16 f2bf(float f) {
    unsigned u = __builtin_bit_cast(unsigned, f);
    u = (u + 0x7FFFu + ((u >> 16) & 1u)) >> 16;          // round-to-nearest-even
    unsigned short h = (unsigned short)u;
    return __builtin_bit_cast(__bf16, h);
}

#define CAT16(lo, hi) __builtin_shufflevector(lo, hi, 0,1,2,3,4,5,6,7,8,9,10,11,12,13,14,15)
#define WMMA_BF16(a, b, c) \
    __builtin_amdgcn_wmma_f32_16x16x32_bf16(false, (a), false, (b), (short)0, (c), false, false)

// ---------------------------------------------------------------------------
// Kernel 0a: X f32 -> bf16, done once so every GEMM pass moves half the bytes.
// ---------------------------------------------------------------------------
__global__ __launch_bounds__(256)
void xcast_kernel(const float* __restrict__ X, __bf16* __restrict__ Xb) {
    const size_t i0 = ((size_t)blockIdx.x * 256 + threadIdx.x) * 8;
#pragma unroll
    for (int e = 0; e < 8; ++e) Xb[i0 + e] = f2bf(X[i0 + e]);
}

// ---------------------------------------------------------------------------
// Kernel 0b: pre-swizzle Wq/Wk/Wv into WMMA B-fragment order:
//   frag f = kstep*NT + ntile;  Wsw[(f*32 + lane)*16 + e] = W[(ks*32+kbB+e)][hcol]
// After this, the GEMM B operand is one aligned 32-byte load per step.
// ---------------------------------------------------------------------------
__global__ __launch_bounds__(32)
void wswz_kernel(const float* __restrict__ Wq, const float* __restrict__ Wk,
                 const float* __restrict__ Wv,
                 __bf16* __restrict__ WqSw, __bf16* __restrict__ WkSw,
                 __bf16* __restrict__ WvSw) {
    const int ks   = blockIdx.x;                 // 0 .. EMB/32-1
    const int nt   = blockIdx.y;                 // 0 .. NT-1
    const int lane = threadIdx.x;
    const int kbB  = (lane < 16) ? 0 : 16;
    const int hcol = nt * 16 + (lane & 15);
    const size_t o = ((size_t)(ks * NT + nt) * 32 + lane) * 16;
#pragma unroll
    for (int e = 0; e < 16; ++e) {
        const size_t s = (size_t)(ks * 32 + kbB + e) * HEAD + hcol;
        WqSw[o + e] = f2bf(Wq[s]);
        WkSw[o + e] = f2bf(Wk[s]);
        WvSw[o + e] = f2bf(Wv[s]);
    }
}

// ---------------------------------------------------------------------------
// Kernel 1: Q/K/V projections. One 8-wave workgroup per 16-row M-tile.
// The 16x1024 bf16 X slab (32 KB, contiguous) is staged into LDS by the
// Tensor Data Mover (one tensor_load_to_lds issued by wave 0), then each wave
// owns one 16-col N-tile with three WMMA accumulators sharing the A fragment.
// V is written directly in B-fragment order for the P@V pass.
// ---------------------------------------------------------------------------
__global__ __launch_bounds__(256)
void qkv_kernel(const __bf16* __restrict__ Xb,
                const __bf16* __restrict__ WqSw,
                const __bf16* __restrict__ WkSw,
                const __bf16* __restrict__ WvSw,
                __bf16* __restrict__ Qb,
                __bf16* __restrict__ Kb,
                __bf16* __restrict__ Vsw) {
    __shared__ __bf16 sX[16 * EMB];              // 32 KB, LDS offset 0
    const int lane  = threadIdx.x & 31;
    const int nt    = threadIdx.x >> 5;          // N-tile owned by this wave
    const int mtile = blockIdx.x;
    const __bf16* src = Xb + (size_t)mtile * 16 * EMB;

#ifdef USE_TDM
    if (threadIdx.x < 32) {
        const unsigned long long ga = (unsigned long long)(uintptr_t)src;
        // D# group0: count=1, lds_addr=0, 57-bit global addr, type=2 ("image")
        v4u g0 = { 1u, 0u, (unsigned)ga,
                   (unsigned)((ga >> 32) & 0x1FFFFFFu) | (2u << 30) };
        // D# group1: data_size=2B; 1-D tile of 16*1024 elements (32 KB)
        v8i g1 = { (int)(1u << 16),                       // data_size = 1 (2 bytes)
                   (int)(((16u * EMB) & 0xFFFFu) << 16),  // tensor_dim0[15:0]
                   (int)(((16u * EMB) >> 16) | (1u << 16)), // tensor_dim0 hi | tensor_dim1=1
                   (int)((16u * EMB) << 16),              // tile_dim0 = 16384
                   0,                                     // tile_dim1 = 0 (unused)
                   (int)(16u * EMB),                      // tensor_dim0_stride
                   0, 0 };
        v4i gz = { 0, 0, 0, 0 };
#if __clang_major__ >= 23
        v8i gz8 = { 0, 0, 0, 0, 0, 0, 0, 0 };
        __builtin_amdgcn_tensor_load_to_lds(g0, g1, gz, gz, gz8, 0);
#else
        __builtin_amdgcn_tensor_load_to_lds(g0, g1, gz, gz, 0);
#endif
        __builtin_amdgcn_s_wait_tensorcnt(0);
    }
    __syncthreads();
#else
    {   // fallback: cooperative wide copy
        const v8bf* s8 = (const v8bf*)src;
        v8bf*       d8 = (v8bf*)sX;
        for (int i = threadIdx.x; i < 16 * EMB / 8; i += 256) d8[i] = s8[i];
        __syncthreads();
    }
#endif

    const int m    = lane & 15;
    const int kloA = (lane < 16) ? 0 : 16;
    const int hcol = nt * 16 + m;
    const v16bf* WqF = (const v16bf*)WqSw;
    const v16bf* WkF = (const v16bf*)WkSw;
    const v16bf* WvF = (const v16bf*)WvSw;

    v8f accq = {0.f,0.f,0.f,0.f,0.f,0.f,0.f,0.f};
    v8f acck = accq, accv = accq;

    for (int kk = 0; kk < EMB; kk += 32) {
        const int ks = kk >> 5;
        const v8bf a0 = *(const v8bf*)&sX[m * EMB + kk + kloA];       // ds_load_b128
        const v8bf a1 = *(const v8bf*)&sX[m * EMB + kk + kloA + 8];  // ds_load_b128
        const v16bf a = CAT16(a0, a1);
        const size_t f = (size_t)(ks * NT + nt) * 32 + lane;
        const v16bf bq = WqF[f];
        const v16bf bk = WkF[f];
        const v16bf bv = WvF[f];
        if (kk + 32 < EMB) __builtin_prefetch(&WqF[f + (size_t)NT * 32], 0, 1);
        accq = WMMA_BF16(a, bq, accq);
        acck = WMMA_BF16(a, bk, acck);
        accv = WMMA_BF16(a, bv, accv);
    }

    const int mo    = (lane < 16) ? 0 : 8;
    const int batch = mtile / (TSEQ / 16);
    const int rb0   = (mtile % (TSEQ / 16)) * 16;
    __bf16* VswB = Vsw + (size_t)batch * TSEQ * HEAD;
#pragma unroll
    for (int r = 0; r < 8; ++r) {
        const int rowg = mtile * 16 + r + mo;
        Qb[(size_t)rowg * HEAD + hcol] = f2bf(accq[r]);
        Kb[(size_t)rowg * HEAD + hcol] = f2bf(acck[r]);
        // V in B-fragment order: rb -> (kstep, lane-half, elem)
        const int rb = rb0 + r + mo;
        const int ks = rb >> 5, ek = rb & 31;
        const int fl = m + ((ek & 16) ? 16 : 0);
        VswB[((size_t)(ks * NT + nt) * 32 + fl) * 16 + (ek & 15)] = f2bf(accv[r]);
    }
}

// ---------------------------------------------------------------------------
// Kernel 2: S = (Q K^T)/sqrt(EMB), 32x32 register-blocked (2x2 WMMA tiles),
// lower-triangular blocks only; -inf above the diagonal. Four independent
// accumulator chains hide the WMMA->WMMA hazard latency.
// ---------------------------------------------------------------------------
__global__ __launch_bounds__(32)
void scores_kernel(const __bf16* __restrict__ Qb,
                   const __bf16* __restrict__ Kb,
                   float* __restrict__ S, int b) {
    const int q2 = blockIdx.x;                    // 32-row block
    const int k2 = blockIdx.y;                    // 32-col block
    if (k2 > q2) return;                          // uniform across wave
    const int lane = threadIdx.x;
    const int m    = lane & 15;
    const int kloA = (lane < 16) ? 0 : 16;
    const int kbB  = (lane < 16) ? 0 : 16;

    const __bf16* Q = Qb + (size_t)b * TSEQ * HEAD;
    const __bf16* K = Kb + (size_t)b * TSEQ * HEAD;

    v8f acc[2][2];
#pragma unroll
    for (int i = 0; i < 2; ++i)
#pragma unroll
        for (int j = 0; j < 2; ++j) acc[i][j] = (v8f){0.f,0.f,0.f,0.f,0.f,0.f,0.f,0.f};

    for (int kk = 0; kk < HEAD; kk += 32) {
        const __bf16* qa0 = Q + (size_t)(q2 * 32 + m) * HEAD + kk + kloA;
        const __bf16* qa1 = qa0 + (size_t)16 * HEAD;
        const v16bf aA = CAT16(*(const v8bf*)qa0, *(const v8bf*)(qa0 + 8));
        const v16bf aB = CAT16(*(const v8bf*)qa1, *(const v8bf*)(qa1 + 8));
        const v16bf b0 = *(const v16bf*)(K + (size_t)(k2 * 32 + m)      * HEAD + kk + kbB);
        const v16bf b1 = *(const v16bf*)(K + (size_t)(k2 * 32 + 16 + m) * HEAD + kk + kbB);
        acc[0][0] = WMMA_BF16(aA, b0, acc[0][0]);
        acc[0][1] = WMMA_BF16(aA, b1, acc[0][1]);
        acc[1][0] = WMMA_BF16(aB, b0, acc[1][0]);
        acc[1][1] = WMMA_BF16(aB, b1, acc[1][1]);
    }

    const float scale = 0.03125f;                 // 1/sqrt(1024)
    const int   mo = (lane < 16) ? 0 : 8;
#pragma unroll
    for (int qh = 0; qh < 2; ++qh)
#pragma unroll
        for (int kh = 0; kh < 2; ++kh) {
            const int kg = k2 * 32 + kh * 16 + m;
#pragma unroll
            for (int r = 0; r < 8; ++r) {
                const int qg = q2 * 32 + qh * 16 + r + mo;
                S[(size_t)qg * TSEQ + kg] =
                    (kg <= qg) ? acc[qh][kh][r] * scale : -__builtin_inff();
            }
        }
}

// ---------------------------------------------------------------------------
// Kernel 3: per-key-column softmax stats over the QUERY axis (q in [k, T)).
// ---------------------------------------------------------------------------
__global__ __launch_bounds__(32)
void colstats_kernel(const float* __restrict__ S,
                     float* __restrict__ colmax,
                     float* __restrict__ colrcp) {
    const int k    = blockIdx.x;
    const int lane = threadIdx.x;

    float mx = -__builtin_inff();
    for (int q = k + lane; q < TSEQ; q += 32)
        mx = fmaxf(mx, S[(size_t)q * TSEQ + k]);
#pragma unroll
    for (int o = 16; o > 0; o >>= 1)
        mx = fmaxf(mx, __shfl_xor(mx, o, 32));

    float sm = 0.f;
    for (int q = k + lane; q < TSEQ; q += 32)
        sm += __expf(S[(size_t)q * TSEQ + k] - mx);
#pragma unroll
    for (int o = 16; o > 0; o >>= 1)
        sm += __shfl_xor(sm, o, 32);

    if (lane == 0) { colmax[k] = mx; colrcp[k] = 1.0f / sm; }
}

// ---------------------------------------------------------------------------
// Kernel 4: Out = P @ V. One 8-wave block per q-tile: the block cooperatively
// materializes each 16x32 P tile ONCE into LDS (coalesced S reads, 2 exps per
// thread instead of 16 per lane per h-tile), then every wave builds its A
// fragment with two ds_load_b128 and WMMAs against its own V fragment.
// Double-buffered sP, one barrier per step (__syncthreads waits DScnt).
// ---------------------------------------------------------------------------
__global__ __launch_bounds__(256)
void out_kernel(const float* __restrict__ S,
                const float* __restrict__ colmax,
                const float* __restrict__ colrcp,
                const __bf16* __restrict__ Vsw,
                float* __restrict__ Out, int b) {
    __shared__ __bf16 sP[2][16 * 32];
    const int qtile = blockIdx.x;
    const int tid   = threadIdx.x;
    const int lane  = tid & 31;
    const int ht    = tid >> 5;                   // h-tile owned by this wave
    const int m     = lane & 15;
    const int kloA  = (lane < 16) ? 0 : 16;
    const int hcol  = ht * 16 + m;

    // P-tile producer mapping: 16 threads per row, 2 columns per thread
    const int prow = tid >> 4;                    // 0..15
    const int pc   = (tid & 15) * 2;              // 0,2,..,30
    const int qg_p = qtile * 16 + prow;
    const float* srowP = S + (size_t)qg_p * TSEQ;

    const v16bf* VF = (const v16bf*)(Vsw + (size_t)b * TSEQ * HEAD);
    float*       O  = Out + (size_t)b * TSEQ * HEAD;

    v8f acc = {0.f,0.f,0.f,0.f,0.f,0.f,0.f,0.f};
    const int kEnd = qtile * 16 + 16;
    int buf = 0;

    for (int kk = 0; kk < kEnd; kk += 32) {
#pragma unroll
        for (int j = 0; j < 2; ++j) {
            const int kg = kk + pc + j;
            const float p = (kg <= qg_p)
                ? __expf(srowP[kg] - colmax[kg]) * colrcp[kg] : 0.0f;
            sP[buf][prow * 32 + pc + j] = f2bf(p);
        }
        __syncthreads();
        const v8bf p0 = *(const v8bf*)&sP[buf][m * 32 + kloA];        // ds_load_b128
        const v8bf p1 = *(const v8bf*)&sP[buf][m * 32 + kloA + 8];   // ds_load_b128
        const v16bf a  = CAT16(p0, p1);
        const v16bf bm = VF[(size_t)((kk >> 5) * NT + ht) * 32 + lane];
        acc = WMMA_BF16(a, bm, acc);
        buf ^= 1;
    }

    const int mo = (lane < 16) ? 0 : 8;
#pragma unroll
    for (int r = 0; r < 8; ++r)
        O[(size_t)(qtile * 16 + r + mo) * HEAD + hcol] = acc[r];
}

// ---------------------------------------------------------------------------
extern "C" void kernel_launch(void* const* d_in, const int* in_sizes, int n_in,
                              void* d_out, int out_size, void* d_ws, size_t ws_size,
                              hipStream_t stream) {
    (void)in_sizes; (void)n_in; (void)out_size; (void)ws_size;
    const float* X  = (const float*)d_in[0];
    const float* Wq = (const float*)d_in[1];
    const float* Wk = (const float*)d_in[2];
    const float* Wv = (const float*)d_in[3];
    float* Out = (float*)d_out;

    // Workspace (~61 MiB): Xb bf16, Qb/Kb bf16, Vsw (fragment order),
    // swizzled weights, S f32 (one batch), column stats.
    __bf16* Xb   = (__bf16*)d_ws;
    __bf16* Qb   = Xb + (size_t)NB * TSEQ * EMB;
    __bf16* Kb   = Qb + (size_t)NB * TSEQ * HEAD;
    __bf16* Vsw  = Kb + (size_t)NB * TSEQ * HEAD;
    __bf16* WqSw = Vsw + (size_t)NB * TSEQ * HEAD;
    __bf16* WkSw = WqSw + (size_t)EMB * HEAD;
    __bf16* WvSw = WkSw + (size_t)EMB * HEAD;
    float*  S    = (float*)(WvSw + (size_t)EMB * HEAD);
    float*  colmax = S + (size_t)TSEQ * TSEQ;
    float*  colrcp = colmax + TSEQ;

    xcast_kernel<<<(NB * TSEQ * EMB) / (256 * 8), 256, 0, stream>>>(X, Xb);

    dim3 gw(EMB / 32, NT);
    wswz_kernel<<<gw, 32, 0, stream>>>(Wq, Wk, Wv, WqSw, WkSw, WvSw);

    qkv_kernel<<<NB * TSEQ / 16, 256, 0, stream>>>(Xb, WqSw, WkSw, WvSw, Qb, Kb, Vsw);

    for (int b = 0; b < NB; ++b) {
        dim3 gsc(TSEQ / 32, TSEQ / 32);
        scores_kernel<<<gsc, 32, 0, stream>>>(Qb, Kb, S, b);
        colstats_kernel<<<TSEQ, 32, 0, stream>>>(S, colmax, colrcp);
        out_kernel<<<TSEQ / 16, 256, 0, stream>>>(S, colmax, colrcp, Vsw, Out, b);
    }
}
